// ComplexMoELayer_3539053051938
// MI455X (gfx1250) — matrix operand order repaired and checked by reading, not compiled, not run on hardware
//
#include <hip/hip_runtime.h>
#include <math.h>

// Problem constants (from reference)
#define B_   2
#define S_   1024
#define F_   512
#define H_   2048
#define E_   8
#define NTOK (B_*S_)          // 2048 tokens
#define ROWCAP (NTOK + 16*E_) // tile-padded staging rows (2176)
#define PI_F 3.14159265358979323846f

typedef __attribute__((ext_vector_type(16))) __bf16          v16bf;
typedef __attribute__((ext_vector_type(16))) unsigned short  v16u;
typedef __attribute__((ext_vector_type(8)))  float           v8f;

union Frag {
    v16u           u;
    v16bf          b;
    unsigned int   d[8];
    unsigned short s[16];
    uint4          q[2];
};

__device__ __forceinline__ unsigned short f2bf(float f) {
    unsigned int x = __builtin_bit_cast(unsigned int, f);
    x += 0x7FFFu + ((x >> 16) & 1u);     // round-to-nearest-even
    return (unsigned short)(x >> 16);
}

// ---------------- workspace layout (bytes) ----------------
static constexpr size_t WS_COUNTS  = 0;                          // 8 * 4
static constexpr size_t WS_OFFP    = 64;                         // 8 * 4 (tile-padded offsets)
static constexpr size_t WS_BUCKETS = 128;                        // E*NTOK*4 = 65536
static constexpr size_t WSZ_W      = (size_t)E_ * F_ * H_ * 2;   // 16,777,216 B each
static constexpr size_t WS_WR1     = 65664;
static constexpr size_t WS_WI1     = WS_WR1 + WSZ_W;
static constexpr size_t WS_WR2     = WS_WI1 + WSZ_W;
static constexpr size_t WS_WI2     = WS_WR2 + WSZ_W;
static constexpr size_t WSZ_XB     = (size_t)ROWCAP * F_ * 2;    // staged bf16 activations
static constexpr size_t WS_XBR     = WS_WI2 + WSZ_W;
static constexpr size_t WS_XBI     = WS_XBR + WSZ_XB;
static constexpr size_t WSZ_H      = (size_t)ROWCAP * H_ * 2;    // staged bf16 hidden
static constexpr size_t WS_HR      = WS_XBI + WSZ_XB;
static constexpr size_t WS_HI      = WS_HR + WSZ_H;

// ---------------- kernels ----------------
__global__ void k_init(int* counts) {
    if (threadIdx.x < E_) counts[threadIdx.x] = 0;
}

// one wave per token: mean-phase routing, bucket append
__global__ __launch_bounds__(256) void k_route(const float* __restrict__ xr,
                                               const float* __restrict__ xi,
                                               int* counts, int* buckets) {
    int wave = threadIdx.x >> 5;
    int lane = threadIdx.x & 31;
    int tok  = blockIdx.x * (blockDim.x >> 5) + wave;
    if (tok >= NTOK) return;
    const float* r = xr + (size_t)tok * F_;
    const float* i = xi + (size_t)tok * F_;
    float sc = 0.f, ss = 0.f;
    for (int f = lane; f < F_; f += 32) {
        float a = r[f], b = i[f];
        float inv = rsqrtf(a * a + b * b);   // cos/sin of atan2(b,a)
        sc += a * inv;
        ss += b * inv;
    }
    #pragma unroll
    for (int o = 16; o > 0; o >>= 1) {
        sc += __shfl_xor(sc, o, 32);
        ss += __shfl_xor(ss, o, 32);
    }
    if (lane == 0) {
        float ph   = atan2f(ss, sc);
        float norm = (ph + PI_F) * (0.5f / PI_F);
        int idx = (int)floorf(norm * (float)E_);
        idx = idx < 0 ? 0 : (idx > E_ - 1 ? E_ - 1 : idx);
        int slot = atomicAdd(&counts[idx], 1);
        buckets[idx * NTOK + slot] = tok;
    }
}

// tile-padded exclusive scan of per-expert counts
__global__ void k_scan(const int* counts, int* offP) {
    if (threadIdx.x == 0 && blockIdx.x == 0) {
        int acc = 0;
        for (int e = 0; e < E_; ++e) {
            offP[e] = acc;
            acc += (counts[e] + 15) & ~15;
        }
    }
}

// fp32 -> bf16 weight conversion, 4 elements/thread
__global__ __launch_bounds__(256) void k_cvt(const float* __restrict__ src,
                                             unsigned short* __restrict__ dst, int n4) {
    int t = blockIdx.x * blockDim.x + threadIdx.x;
    if (t >= n4) return;
    float4 v = ((const float4*)src)[t];
    ushort4 o;
    o.x = f2bf(v.x); o.y = f2bf(v.y); o.z = f2bf(v.z); o.w = f2bf(v.w);
    ((ushort4*)dst)[t] = o;
}

// gather routed tokens into compact, tile-padded bf16 staging; zero tail rows
__global__ __launch_bounds__(256) void k_gather(
    const float* __restrict__ xr, const float* __restrict__ xi,
    const int* __restrict__ counts, const int* __restrict__ offP,
    const int* __restrict__ buckets,
    unsigned short* __restrict__ xbR, unsigned short* __restrict__ xbI,
    unsigned short* __restrict__ hR,  unsigned short* __restrict__ hI)
{
    int m   = blockIdx.x;
    int e   = blockIdx.y;
    int cnt = counts[e];
    int pad = (cnt + 15) & ~15;
    if (m >= pad) return;
    size_t srow = (size_t)(offP[e] + m);
    int t = threadIdx.x;
    if (m < cnt) {
        int tok = buckets[e * NTOK + m];
        const float2* rr = (const float2*)(xr + (size_t)tok * F_);
        const float2* ii = (const float2*)(xi + (size_t)tok * F_);
        // 256 threads x 2 elements = 512
        float2 a = rr[t], b = ii[t];
        ushort2 oa, ob;
        oa.x = f2bf(a.x); oa.y = f2bf(a.y);
        ob.x = f2bf(b.x); ob.y = f2bf(b.y);
        ((ushort2*)(xbR + srow * F_))[t] = oa;
        ((ushort2*)(xbI + srow * F_))[t] = ob;
    } else {
        // zero-fill padded tail rows so partial tiles never feed garbage/NaN
        ((ushort2*)(xbR + srow * F_))[t] = make_ushort2(0, 0);
        ((ushort2*)(xbI + srow * F_))[t] = make_ushort2(0, 0);
        ushort4 z = make_ushort4(0, 0, 0, 0);
        ushort4* zr = (ushort4*)(hR + srow * H_);
        ushort4* zi = (ushort4*)(hI + srow * H_);
        #pragma unroll
        for (int c = 0; c < 2; ++c) {         // 256 thr x 2 x 4 = 2048
            zr[t + c * 256] = z;
            zi[t + c * 256] = z;
        }
    }
}

// layer1: staged bf16 A, complex-GEMM (F x H) + bias + ModReLU -> bf16 staging
__global__ __launch_bounds__(256) void k_gemm1(
    const unsigned short* __restrict__ xbR, const unsigned short* __restrict__ xbI,
    const unsigned short* __restrict__ Wr1b, const unsigned short* __restrict__ Wi1b,
    const float* __restrict__ br1, const float* __restrict__ bi1,
    const float* __restrict__ mb,
    const int* __restrict__ counts, const int* __restrict__ offP,
    unsigned short* __restrict__ hR, unsigned short* __restrict__ hI)
{
    int e     = blockIdx.z;
    int cnt   = counts[e];
    int mtile = blockIdx.y;
    if (mtile * 16 >= cnt) return;                 // block-uniform: EXEC stays full
    int wave = threadIdx.x >> 5, lane = threadIdx.x & 31;
    int n0   = (blockIdx.x * 8 + wave) * 16;
    int lrow = lane & 15, half = lane >> 4;

    size_t srow = (size_t)(offP[e] + mtile * 16 + lrow);
    const unsigned short* aRr = xbR + srow * F_;
    const unsigned short* aIr = xbI + srow * F_;
    const unsigned short* bRp = Wr1b + (size_t)e * F_ * H_ + n0;
    const unsigned short* bIp = Wi1b + (size_t)e * F_ * H_ + n0;

    v8f accR = {}, accI = {};
    for (int k0 = 0; k0 < F_; k0 += 32) {
        int ka = k0 + 8 * half, kb = ka + 16;
        Frag ar, ai, ain, br_, bi_;
        ar.q[0] = *(const uint4*)(aRr + ka);
        ar.q[1] = *(const uint4*)(aRr + kb);
        ai.q[0] = *(const uint4*)(aIr + ka);
        ai.q[1] = *(const uint4*)(aIr + kb);
        #pragma unroll
        for (int j = 0; j < 8; ++j) ain.d[j] = ai.d[j] ^ 0x80008000u;   // -x_i
        #pragma unroll
        for (int s = 0; s < 16; ++s) {
            size_t kk = (size_t)(k0 + 8 * half + (s & 7) + 16 * (s >> 3));
            br_.s[s] = bRp[kk * H_ + lrow];
            bi_.s[s] = bIp[kk * H_ + lrow];
        }
        accR = __builtin_amdgcn_wmma_f32_16x16x32_bf16(false, ar.b,  false, br_.b, (short)0, accR, false, false);
        accR = __builtin_amdgcn_wmma_f32_16x16x32_bf16(false, ain.b, false, bi_.b, (short)0, accR, false, false);
        accI = __builtin_amdgcn_wmma_f32_16x16x32_bf16(false, ar.b,  false, bi_.b, (short)0, accI, false, false);
        accI = __builtin_amdgcn_wmma_f32_16x16x32_bf16(false, ai.b,  false, br_.b, (short)0, accI, false, false);
    }

    int n = n0 + lrow;
    float b1 = br1[e * H_ + n], b2 = bi1[e * H_ + n], mbv = mb[e * H_ + n];
    int off = offP[e];
    #pragma unroll
    for (int j = 0; j < 8; ++j) {
        int m = mtile * 16 + j + 8 * half;
        if (m < cnt) {
            float hr  = accR[j] + b1, hi = accI[j] + b2;
            float amp = sqrtf(hr * hr + hi * hi + 1e-10f);
            float scl = fmaxf(amp + mbv, 0.f) / (amp + 1e-10f);
            size_t o  = (size_t)(off + m) * H_ + n;
            hR[o] = f2bf(scl * hr);
            hI[o] = f2bf(scl * hi);
        }
    }
}

// layer2: h(Ne x H) complex-GEMM (H x F) + bias, scatter to out by token id
__global__ __launch_bounds__(256) void k_gemm2(
    const unsigned short* __restrict__ hR, const unsigned short* __restrict__ hI,
    const unsigned short* __restrict__ Wr2b, const unsigned short* __restrict__ Wi2b,
    const float* __restrict__ br2, const float* __restrict__ bi2,
    const int* __restrict__ counts, const int* __restrict__ offP,
    const int* __restrict__ buckets,
    float* __restrict__ outR, float* __restrict__ outI)
{
    int e     = blockIdx.z;
    int cnt   = counts[e];
    int mtile = blockIdx.y;
    if (mtile * 16 >= cnt) return;
    int wave = threadIdx.x >> 5, lane = threadIdx.x & 31;
    int n0   = (blockIdx.x * 8 + wave) * 16;
    int lrow = lane & 15, half = lane >> 4;

    size_t srow = (size_t)(offP[e] + mtile * 16 + lrow);
    const unsigned short* hRr = hR + srow * H_;
    const unsigned short* hIr = hI + srow * H_;
    const unsigned short* bRp = Wr2b + (size_t)e * H_ * F_ + n0;
    const unsigned short* bIp = Wi2b + (size_t)e * H_ * F_ + n0;

    v8f accR = {}, accI = {};
    for (int k0 = 0; k0 < H_; k0 += 32) {
        int ka = k0 + 8 * half, kb = ka + 16;
        Frag ar, ai, ain, br_, bi_;
        ar.q[0] = *(const uint4*)(hRr + ka);
        ar.q[1] = *(const uint4*)(hRr + kb);
        ai.q[0] = *(const uint4*)(hIr + ka);
        ai.q[1] = *(const uint4*)(hIr + kb);
        #pragma unroll
        for (int j = 0; j < 8; ++j) ain.d[j] = ai.d[j] ^ 0x80008000u;   // -h_i
        #pragma unroll
        for (int s = 0; s < 16; ++s) {
            size_t kk = (size_t)(k0 + 8 * half + (s & 7) + 16 * (s >> 3));
            br_.s[s] = bRp[kk * F_ + lrow];
            bi_.s[s] = bIp[kk * F_ + lrow];
        }
        accR = __builtin_amdgcn_wmma_f32_16x16x32_bf16(false, ar.b,  false, br_.b, (short)0, accR, false, false);
        accR = __builtin_amdgcn_wmma_f32_16x16x32_bf16(false, ain.b, false, bi_.b, (short)0, accR, false, false);
        accI = __builtin_amdgcn_wmma_f32_16x16x32_bf16(false, ar.b,  false, bi_.b, (short)0, accI, false, false);
        accI = __builtin_amdgcn_wmma_f32_16x16x32_bf16(false, ai.b,  false, br_.b, (short)0, accI, false, false);
    }

    int n = n0 + lrow;
    float b1 = br2[e * F_ + n], b2 = bi2[e * F_ + n];
    #pragma unroll
    for (int j = 0; j < 8; ++j) {
        int m = mtile * 16 + j + 8 * half;
        if (m < cnt) {
            int t = buckets[e * NTOK + m];
            outR[(size_t)t * F_ + n] = accR[j] + b1;
            outI[(size_t)t * F_ + n] = accI[j] + b2;
        }
    }
}

// ---------------- launcher ----------------
extern "C" void kernel_launch(void* const* d_in, const int* in_sizes, int n_in,
                              void* d_out, int out_size, void* d_ws, size_t ws_size,
                              hipStream_t stream) {
    const float* xr  = (const float*)d_in[0];
    const float* xi  = (const float*)d_in[1];
    const float* Wr1 = (const float*)d_in[2];
    const float* Wi1 = (const float*)d_in[3];
    const float* br1 = (const float*)d_in[4];
    const float* bi1 = (const float*)d_in[5];
    const float* mb  = (const float*)d_in[6];
    const float* Wr2 = (const float*)d_in[7];
    const float* Wi2 = (const float*)d_in[8];
    const float* br2 = (const float*)d_in[9];
    const float* bi2 = (const float*)d_in[10];

    char* ws = (char*)d_ws;
    int* counts  = (int*)(ws + WS_COUNTS);
    int* offP    = (int*)(ws + WS_OFFP);
    int* buckets = (int*)(ws + WS_BUCKETS);
    unsigned short* Wr1b = (unsigned short*)(ws + WS_WR1);
    unsigned short* Wi1b = (unsigned short*)(ws + WS_WI1);
    unsigned short* Wr2b = (unsigned short*)(ws + WS_WR2);
    unsigned short* Wi2b = (unsigned short*)(ws + WS_WI2);
    unsigned short* xbR  = (unsigned short*)(ws + WS_XBR);
    unsigned short* xbI  = (unsigned short*)(ws + WS_XBI);
    unsigned short* hR   = (unsigned short*)(ws + WS_HR);
    unsigned short* hI   = (unsigned short*)(ws + WS_HI);

    float* outR = (float*)d_out;
    float* outI = outR + (size_t)B_ * S_ * F_;

    k_init<<<1, 32, 0, stream>>>(counts);
    k_route<<<NTOK / 8, 256, 0, stream>>>(xr, xi, counts, buckets);
    k_scan<<<1, 32, 0, stream>>>(counts, offP);

    int n4 = (E_ * F_ * H_) / 4;                       // 2,097,152 threads
    k_cvt<<<n4 / 256, 256, 0, stream>>>(Wr1, Wr1b, n4);
    k_cvt<<<n4 / 256, 256, 0, stream>>>(Wi1, Wi1b, n4);
    k_cvt<<<n4 / 256, 256, 0, stream>>>(Wr2, Wr2b, n4);
    k_cvt<<<n4 / 256, 256, 0, stream>>>(Wi2, Wi2b, n4);

    k_gather<<<dim3(NTOK, E_), 256, 0, stream>>>(xr, xi, counts, offP, buckets,
                                                 xbR, xbI, hR, hI);

    dim3 g1(H_ / 128, NTOK / 16, E_);                  // (16, 128, 8)
    k_gemm1<<<g1, 256, 0, stream>>>(xbR, xbI, Wr1b, Wi1b, br1, bi1, mb,
                                    counts, offP, hR, hI);

    dim3 g2(F_ / 128, NTOK / 16, E_);                  // (4, 128, 8)
    k_gemm2<<<g2, 256, 0, stream>>>(hR, hI, Wr2b, Wi2b, br2, bi2,
                                    counts, offP, buckets, outR, outI);
}
